// MultiModalNetTextualKernels_NoVisual_300647711058
// MI455X (gfx1250) — compile-verified, matrix operand order checked by reading, not run on hardware
//
#include <hip/hip_runtime.h>
#include <hip/hip_bf16.h>
#include <stdint.h>

// ---------------------------------------------------------------------------
// MI455X (gfx1250, wave32) implementation. All heavy GEMMs/convs run on
// v_wmma_f32_16x16x32_bf16. Activations live as [pixel][channel] bf16 with
// channel counts padded to multiples of 32. Conv weights are staged into LDS
// by the Tensor Data Mover (tensor_load_to_lds) when the builtin is available.
// ---------------------------------------------------------------------------

typedef unsigned short ushort_t;
typedef __attribute__((ext_vector_type(16))) __bf16 v16bf;
typedef __attribute__((ext_vector_type(8)))  float  v8f;
typedef __attribute__((ext_vector_type(4)))  unsigned int u32x4;
typedef __attribute__((ext_vector_type(8)))  int i32x8;
typedef __attribute__((ext_vector_type(4)))  int i32x4;

union ABf { v16bf v; uint4 q[2]; };

#if defined(__has_builtin)
#if __has_builtin(__builtin_amdgcn_tensor_load_to_lds) && \
    __has_builtin(__builtin_amdgcn_s_wait_tensorcnt)
#define HAVE_TDM 1
#endif
#endif
#ifndef HAVE_TDM
#define HAVE_TDM 0
#endif

__device__ __forceinline__ ushort_t f2bf(float f) {
  unsigned u = __builtin_bit_cast(unsigned, f);
  unsigned r = u + 0x7FFFu + ((u >> 16) & 1u);
  return (ushort_t)(r >> 16);
}

#define BN_EPS 1e-3f

#if HAVE_TDM
// Issue a TDM 2-D tile load: rows x width bf16 elements, global row stride
// rowStride (elements), into contiguous LDS at ldsAddr. Wave-level op.
__device__ __forceinline__ void tdm_load_tile(const ushort_t* gsrc,
                                              unsigned ldsAddr,
                                              unsigned width, unsigned rows,
                                              unsigned long long rowStride) {
  unsigned long long ga = (unsigned long long)(uintptr_t)gsrc;
  unsigned long long strideBytesElems = rowStride;   // in data_size units
  u32x4 g0;
  g0.x = 1u;                                   // count=1, user descriptor
  g0.y = ldsAddr;                              // lds_addr (bytes)
  g0.z = (unsigned)(ga & 0xFFFFFFFFull);       // global_addr[31:0]
  g0.w = (unsigned)((ga >> 32) & 0x01FFFFFFull) | (2u << 30); // addr hi | type=2
  i32x8 g1;
  g1[0] = (int)(1u << 16);                     // data_size = 1 (2 bytes)
  g1[1] = (int)((width & 0xFFFFu) << 16);      // tensor_dim0[15:0] @ bits 63:48
  g1[2] = (int)(((width >> 16) & 0xFFFFu) | ((rows & 0xFFFFu) << 16)); // d0 hi | d1 lo
  g1[3] = (int)(((rows >> 16) & 0xFFFFu) | ((width & 0xFFFFu) << 16)); // d1 hi | tile_dim0
  g1[4] = (int)(rows & 0xFFFFu);               // tile_dim1 | tile_dim2=0
  g1[5] = (int)(strideBytesElems & 0xFFFFFFFFull);        // dim0_stride lo32
  g1[6] = (int)((strideBytesElems >> 32) & 0xFFFFull);    // dim0_stride hi16
  g1[7] = 0;
  i32x4 gz = {0, 0, 0, 0};
#if __clang_major__ >= 23
  i32x8 gz8 = {0, 0, 0, 0, 0, 0, 0, 0};
  __builtin_amdgcn_tensor_load_to_lds(g0, g1, gz, gz, gz8, 0);
#else
  __builtin_amdgcn_tensor_load_to_lds(g0, g1, gz, gz, 0);
#endif
}
#endif

// ---------------------------------------------------------------------------
// Kernel: fc banks (tweet 20x + img 5x reused twice) -> kern bf16 [B][32][2048]
// ---------------------------------------------------------------------------
__global__ void fc_bank_kernel(
    const float* __restrict__ x2, const float* __restrict__ x3,
    const float* __restrict__ twW, const float* __restrict__ twBi,
    const float* __restrict__ twBe, const float* __restrict__ twG,
    const float* __restrict__ twMu, const float* __restrict__ twVa,
    const float* __restrict__ imW, const float* __restrict__ imBi,
    const float* __restrict__ imBe, const float* __restrict__ imG,
    const float* __restrict__ imMu, const float* __restrict__ imVa,
    ushort_t* __restrict__ kern)
{
  int idx = blockIdx.x * 256 + threadIdx.x;
  if (idx >= 256 * 32 * 2048) return;
  int co = idx & 2047, k = (idx >> 11) & 31, b = idx >> 16;
  if (k >= 30) { kern[idx] = 0; return; }
  const float* w; const float* x; int pi; float bias, g, be, mu, va;
  if (k < 20) {
    pi = k * 2048 + co; w = twW + (size_t)pi * 50; x = x3 + b * 50;
    bias = twBi[pi]; g = twG[pi]; be = twBe[pi]; mu = twMu[pi]; va = twVa[pi];
  } else {
    int ki = (k - 20) % 5;  // img bank reused for k6..k10 (source bug preserved)
    pi = ki * 2048 + co; w = imW + (size_t)pi * 50; x = x2 + b * 50;
    bias = imBi[pi]; g = imG[pi]; be = imBe[pi]; mu = imMu[pi]; va = imVa[pi];
  }
  float acc = bias;
  #pragma unroll
  for (int i = 0; i < 50; ++i) acc += w[i] * x[i];
  float sc = g * rsqrtf(va + BN_EPS);
  kern[idx] = f2bf(fmaxf(acc * sc + (be - mu * sc), 0.f));
}

// ---------------------------------------------------------------------------
// Kernel: x1 [B][2048][64] f32 -> x1t [B*64][2048] bf16 (LDS-tiled transpose)
// ---------------------------------------------------------------------------
__global__ __launch_bounds__(256) void x1_transpose_kernel(
    const float* __restrict__ x1, ushort_t* __restrict__ x1t)
{
  __shared__ float tile[32][65];
  int b = blockIdx.y, c0 = blockIdx.x * 32, tid = threadIdx.x;
  for (int i = tid; i < 2048; i += 256) {
    int c = i >> 6, p = i & 63;
    tile[c][p] = x1[((size_t)b * 2048 + c0 + c) * 64 + p];
  }
  __syncthreads();
  for (int i = tid; i < 2048; i += 256) {
    int c = i & 31, p = i >> 5;
    x1t[((size_t)b * 64 + p) * 2048 + c0 + c] = f2bf(tile[c][p]);
  }
}

// ---------------------------------------------------------------------------
// Kernel: fill FEAT channels 30..159  (FEAT: [N=16384][160] bf16)
// ---------------------------------------------------------------------------
__global__ void feat_fill_kernel(const float* __restrict__ x2,
                                 const float* __restrict__ x3,
                                 ushort_t* __restrict__ feat)
{
  int idx = blockIdx.x * 256 + threadIdx.x;
  if (idx >= 16384 * 130) return;
  int j = idx % 130, n = idx / 130, b = n >> 6;
  int c = 30 + j;
  float v = 0.f;
  if (c < 80) v = x2[b * 50 + (c - 30)];
  else if (c < 130) v = x3[b * 50 + (c - 80)];
  feat[(size_t)n * 160 + c] = f2bf(v);
}

// ---------------------------------------------------------------------------
// Kernel: weight convert f32 [co][ci][tapsSrc] -> bf16 [co][tapsDst][cinPad]
// ---------------------------------------------------------------------------
__global__ void w2bf_kernel(const float* __restrict__ src, ushort_t* __restrict__ dst,
                            int cin, int cinPad, int tapsSrc, int tapsDst,
                            float scale, int total)
{
  int idx = blockIdx.x * 256 + threadIdx.x;
  if (idx >= total) return;
  int ci = idx % cinPad;
  int rest = idx / cinPad;
  int t = rest % tapsDst;
  int co = rest / tapsDst;
  float v = 0.f;
  if (ci < cin) {
    int ts = (tapsSrc == tapsDst) ? t : 0;
    v = src[((size_t)co * cin + ci) * tapsSrc + ts] * scale;
  }
  dst[idx] = f2bf(v);
}

// ---------------------------------------------------------------------------
// Kernel: mm einsum out[b,k,p] = sum_c kern[b,k,c] * x1[b,c,p] via WMMA.
// ---------------------------------------------------------------------------
__global__ __launch_bounds__(256) void mm_wmma_kernel(
    const ushort_t* __restrict__ kern,   // [B][32][2048]
    const ushort_t* __restrict__ x1t,    // [N][2048]
    const float* __restrict__ g, const float* __restrict__ be,
    const float* __restrict__ mu, const float* __restrict__ va,   // [30]
    ushort_t* __restrict__ feat)         // [N][160]
{
  __shared__ ushort_t smA[32 * 256];
  const int tid = threadIdx.x, wave = tid >> 5, lane = tid & 31;
  const int m = lane & 15, half = lane >> 4;
  const int b = blockIdx.x;
  const int mtile = wave >> 2, ntile = wave & 3;
  const int n = b * 64 + ntile * 16 + m;
  const ushort_t* abase = kern + (size_t)b * 32 * 2048;

  v8f acc = {0.f,0.f,0.f,0.f,0.f,0.f,0.f,0.f};
  for (int kc = 0; kc < 2048; kc += 256) {
    __syncthreads();
    for (int v = tid; v < 1024; v += 256) {          // 32 rows x 32 vec8
      int row = v >> 5, kv = (v & 31) << 3;
      *(uint4*)(&smA[row * 256 + kv]) =
          *(const uint4*)(abase + (size_t)row * 2048 + kc + kv);
    }
    __syncthreads();
    for (int s8 = 0; s8 < 8; ++s8) {
      ABf A, B;
      const ushort_t* ap = &smA[(mtile * 16 + m) * 256 + (s8 << 5)];
      A.q[0] = *(const uint4*)(ap + 8 * half);
      A.q[1] = *(const uint4*)(ap + 16 + 8 * half);
      const ushort_t* bp = x1t + (size_t)n * 2048 + kc + (s8 << 5);
      B.q[0] = *(const uint4*)(bp + 8 * half);
      B.q[1] = *(const uint4*)(bp + 16 + 8 * half);
      acc = __builtin_amdgcn_wmma_f32_16x16x32_bf16(
          false, A.v, false, B.v, (short)0, acc, false, false);
    }
  }
  int cbase = mtile * 16 + 8 * half;
  #pragma unroll
  for (int r = 0; r < 8; ++r) {
    int c = cbase + r;
    if (c < 30) {
      float sc = g[c] * rsqrtf(va[c] + BN_EPS);
      float y = fmaxf(acc[r] * sc + (be[c] - mu[c] * sc), 0.f);
      feat[(size_t)n * 160 + c] = f2bf(y);
    }
  }
}

// ---------------------------------------------------------------------------
// Kernel: implicit-GEMM conv via WMMA bf16.
// MODE: 0=1x1, 1=1x3, 2=3x1, 3=3x3 (also avg-pool branch, weights pre-/9).
// One wave = 32 cout x 64 px (one image): two A fragments share every B
// fragment -> 8 WMMAs per 32-K step. The 32 x kchunk weight tile is staged
// into LDS by the Tensor Data Mover (wave 0 issues, s_wait_tensorcnt, block
// barrier); falls back to cooperative b128 loads when TDM is unavailable.
// Output: BN+ReLU then either bf16 [n][outStride]@coBase, or (pooled!=null)
// the 8x8 spatial mean reduced in-register straight into pooled[b][2048].
// ---------------------------------------------------------------------------
template <int MODE>
__global__ __launch_bounds__(256) void conv_wmma_kernel(
    const ushort_t* __restrict__ act, int cinPad,
    const ushort_t* __restrict__ wgt,
    const float* __restrict__ bnG, const float* __restrict__ bnB,
    const float* __restrict__ bnM, const float* __restrict__ bnV,
    ushort_t* __restrict__ out, int outStride, int coBase,
    float* __restrict__ pooled, int kchunk)
{
  constexpr int NTAP = (MODE == 0) ? 1 : ((MODE == 3) ? 9 : 3);
  __shared__ ushort_t smA[32 * 256];     // 16 KB: 32 cout rows x kchunk

  const int tid = threadIdx.x;
  const int wave = tid >> 5, lane = tid & 31;
  const int m = lane & 15, half = lane >> 4;
  const int co0 = blockIdx.x * 32;
  const int b = blockIdx.y * 8 + wave;

  // Per-lane source pixel index for each tap / N-subtile (-1 = zero padding).
  int srcOff[NTAP][4];
  #pragma unroll
  for (int t = 0; t < NTAP; ++t) {
    const int dh = (MODE == 2) ? (t - 1) : ((MODE == 3) ? (t / 3 - 1) : 0);
    const int dw = (MODE == 1) ? (t - 1) : ((MODE == 3) ? (t % 3 - 1) : 0);
    #pragma unroll
    for (int s = 0; s < 4; ++s) {
      int p = s * 16 + m, h = p >> 3, w = p & 7;
      int hh = h + dh, ww = w + dw;
      srcOff[t][s] = (hh >= 0 && hh < 8 && ww >= 0 && ww < 8)
                         ? (b * 64 + hh * 8 + ww) : -1;
    }
  }

  v8f zero8 = {0.f,0.f,0.f,0.f,0.f,0.f,0.f,0.f};
  v8f acc[2][4];
  #pragma unroll
  for (int hh = 0; hh < 2; ++hh)
    #pragma unroll
    for (int s = 0; s < 4; ++s) acc[hh][s] = zero8;

  const int nsteps = kchunk >> 5;
  const size_t wRow = (size_t)NTAP * cinPad;
#if HAVE_TDM
  const unsigned ldsAddr = (unsigned)(uintptr_t)(&smA[0]);  // LDS aperture low bits
#else
  const int vpr = kchunk >> 3;      // vec8 per weight row
  const int nvec = vpr * 32;
#endif

  #pragma unroll
  for (int t = 0; t < NTAP; ++t) {
    for (int kc = 0; kc < cinPad; kc += kchunk) {
      __syncthreads();                    // previous smA fully consumed
#if HAVE_TDM
      if (wave == 0) {
        tdm_load_tile(wgt + (size_t)co0 * wRow + (size_t)t * cinPad + kc,
                      ldsAddr, (unsigned)kchunk, 32u, wRow);
        __builtin_amdgcn_s_wait_tensorcnt(0);
      }
#else
      for (int v = tid; v < nvec; v += 256) {
        int co = v / vpr;
        int kv = (v - co * vpr) << 3;
        *(uint4*)(&smA[co * kchunk + kv]) =
            *(const uint4*)(wgt + (size_t)(co0 + co) * wRow +
                            (size_t)t * cinPad + kc + kv);
      }
#endif
      if (tid == 0) {   // warm L2 with the next weight chunk
        size_t nxt = (size_t)t * cinPad + kc + kchunk;
        if (nxt < wRow)
          __builtin_prefetch(wgt + (size_t)co0 * wRow + nxt, 0, 1);
      }
      __syncthreads();
      for (int s8 = 0; s8 < nsteps; ++s8) {
        ABf A0, A1;
        const ushort_t* ap0 = &smA[m * kchunk + (s8 << 5)];
        const ushort_t* ap1 = &smA[(16 + m) * kchunk + (s8 << 5)];
        A0.q[0] = *(const uint4*)(ap0 + 8 * half);
        A0.q[1] = *(const uint4*)(ap0 + 16 + 8 * half);
        A1.q[0] = *(const uint4*)(ap1 + 8 * half);
        A1.q[1] = *(const uint4*)(ap1 + 16 + 8 * half);
        #pragma unroll
        for (int s = 0; s < 4; ++s) {
          ABf B;
          const int sn = srcOff[t][s];
          if (sn >= 0) {
            const ushort_t* bp = act + (size_t)sn * cinPad + kc + (s8 << 5);
            B.q[0] = *(const uint4*)(bp + 8 * half);
            B.q[1] = *(const uint4*)(bp + 16 + 8 * half);
          } else {
            B.q[0] = make_uint4(0u, 0u, 0u, 0u);
            B.q[1] = make_uint4(0u, 0u, 0u, 0u);
          }
          // Divergence only gates the loads; EXEC is all-ones at the WMMAs.
          acc[0][s] = __builtin_amdgcn_wmma_f32_16x16x32_bf16(
              false, A0.v, false, B.v, (short)0, acc[0][s], false, false);
          acc[1][s] = __builtin_amdgcn_wmma_f32_16x16x32_bf16(
              false, A1.v, false, B.v, (short)0, acc[1][s], false, false);
        }
      }
    }
  }

  // Epilogue: fold BN, ReLU; store bf16 NHWC or fused global-average pool.
  #pragma unroll
  for (int hh = 0; hh < 2; ++hh) {
    const int cbase = co0 + hh * 16 + 8 * half;
    float sc[8], sh[8];
    #pragma unroll
    for (int r = 0; r < 8; ++r) {
      float g = bnG[cbase + r], be = bnB[cbase + r];
      float mu = bnM[cbase + r], va = bnV[cbase + r];
      sc[r] = g * rsqrtf(va + BN_EPS);
      sh[r] = be - mu * sc[r];
    }
    if (pooled == nullptr) {
      #pragma unroll
      for (int s = 0; s < 4; ++s) {
        int n = b * 64 + s * 16 + m;
        ushort_t v8[8];
        #pragma unroll
        for (int r = 0; r < 8; ++r)
          v8[r] = f2bf(fmaxf(acc[hh][s][r] * sc[r] + sh[r], 0.f));
        uint4 pk;
        pk.x = (unsigned)v8[0] | ((unsigned)v8[1] << 16);
        pk.y = (unsigned)v8[2] | ((unsigned)v8[3] << 16);
        pk.z = (unsigned)v8[4] | ((unsigned)v8[5] << 16);
        pk.w = (unsigned)v8[6] | ((unsigned)v8[7] << 16);
        *(uint4*)(out + (size_t)n * outStride + coBase + cbase) = pk;
      }
    } else {
      float sum[8];
      #pragma unroll
      for (int r = 0; r < 8; ++r) sum[r] = 0.f;
      #pragma unroll
      for (int s = 0; s < 4; ++s)
        #pragma unroll
        for (int r = 0; r < 8; ++r)
          sum[r] += fmaxf(acc[hh][s][r] * sc[r] + sh[r], 0.f);
      #pragma unroll
      for (int r = 0; r < 8; ++r) {
        sum[r] += __shfl_xor(sum[r], 1, 16);
        sum[r] += __shfl_xor(sum[r], 2, 16);
        sum[r] += __shfl_xor(sum[r], 4, 16);
        sum[r] += __shfl_xor(sum[r], 8, 16);
      }
      if (m == 0) {
        #pragma unroll
        for (int r = 0; r < 8; ++r)
          pooled[(size_t)b * 2048 + coBase + cbase + r] = sum[r] * (1.f / 64.f);
      }
    }
  }
}

// ---------------------------------------------------------------------------
// Kernel: fused FC (+ optional BN+ReLU), naive (tiny fraction of FLOPs).
// ---------------------------------------------------------------------------
__global__ void fc_fused_kernel(const float* __restrict__ x,
                                const float* __restrict__ w,
                                const float* __restrict__ bias,
                                const float* __restrict__ g,
                                const float* __restrict__ be,
                                const float* __restrict__ mu,
                                const float* __restrict__ va,
                                float* __restrict__ y,
                                int cin, int cout, int total, int useBn)
{
  int idx = blockIdx.x * 256 + threadIdx.x;
  if (idx >= total) return;
  int j = idx % cout, b = idx / cout;
  const float* xr = x + (size_t)b * cin;
  const float* wr = w + (size_t)j * cin;
  float acc = bias[j];
  for (int i = 0; i < cin; i += 4)
    acc += wr[i] * xr[i] + wr[i + 1] * xr[i + 1] +
           wr[i + 2] * xr[i + 2] + wr[i + 3] * xr[i + 3];
  if (useBn) {
    float sc = g[j] * rsqrtf(va[j] + BN_EPS);
    acc = fmaxf(acc * sc + (be[j] - mu[j] * sc), 0.f);
  }
  y[idx] = acc;
}

// ---------------------------------------------------------------------------
// Host orchestration
// ---------------------------------------------------------------------------
struct CC { int bnI, wI, cout, cin, cinPad, mode, isPool; };

extern "C" void kernel_launch(void* const* d_in, const int* in_sizes, int n_in,
                              void* d_out, int out_size, void* d_ws, size_t ws_size,
                              hipStream_t stream) {
  (void)in_sizes; (void)n_in; (void)out_size; (void)ws_size;

  // d_in index map (jax sorted-key tree flattening: params first, then x1..x3)
  const float* x1 = (const float*)d_in[120];
  const float* x2 = (const float*)d_in[121];
  const float* x3 = (const float*)d_in[122];

  const int N = 16384;            // 256 images * 64 px
  size_t off = 0;
  auto alloc = [&](size_t bytes) { size_t o = off; off += (bytes + 255) & ~(size_t)255; return o; };
  char* ws = (char*)d_ws;

  ushort_t* Y1   = (ushort_t*)(ws + alloc((size_t)N * 2048 * 2)); // also x1t
  ushort_t* KERN = (ushort_t*)(ws + alloc((size_t)256 * 32 * 2048 * 2));
  ushort_t* FEAT = (ushort_t*)(ws + alloc((size_t)N * 160 * 2));
  ushort_t* T1   = (ushort_t*)(ws + alloc((size_t)N * 384 * 2));
  ushort_t* T2   = (ushort_t*)(ws + alloc((size_t)N * 448 * 2));
  ushort_t* T3   = (ushort_t*)(ws + alloc((size_t)N * 384 * 2));
  float*    POOL = (float*)(ws + alloc((size_t)256 * 2048 * 4));
  float*    H1   = (float*)(ws + alloc((size_t)256 * 1024 * 4));
  float*    H2   = (float*)(ws + alloc((size_t)256 * 512 * 4));

  static const CC ccs[18] = {
    // inc1 (cin 130 -> pad 160)
    {24, 28, 320, 130, 160, 0, 0}, {29, 33, 384, 130, 160, 0, 0},
    {34, 38, 384, 384, 384, 1, 0}, {39, 43, 384, 384, 384, 2, 0},
    {44, 48, 448, 130, 160, 0, 0}, {49, 53, 384, 448, 448, 3, 0},
    {54, 58, 384, 384, 384, 1, 0}, {59, 63, 384, 384, 384, 2, 0},
    {64, 68, 192, 130, 160, 3, 1},
    // inc2 (cin 2048)
    {69, 73, 320, 2048, 2048, 0, 0}, {74, 78, 384, 2048, 2048, 0, 0},
    {79, 83, 384, 384, 384, 1, 0},   {84, 88, 384, 384, 384, 2, 0},
    {89, 93, 448, 2048, 2048, 0, 0}, {94, 98, 384, 448, 448, 3, 0},
    {99, 103, 384, 384, 384, 1, 0},  {104, 108, 384, 384, 384, 2, 0},
    {109, 113, 192, 2048, 2048, 3, 1},
  };

  size_t wbOff[18];
  for (int i = 0; i < 18; ++i) {
    const CC& c = ccs[i];
    int tapsDst = (c.mode == 0) ? 1 : ((c.mode == 3) ? 9 : 3);
    wbOff[i] = alloc((size_t)c.cout * tapsDst * c.cinPad * 2);
  }

  for (int i = 0; i < 18; ++i) {
    const CC& c = ccs[i];
    int tapsDst = (c.mode == 0) ? 1 : ((c.mode == 3) ? 9 : 3);
    int tapsSrc = c.isPool ? 1 : tapsDst;
    float scale = c.isPool ? (1.f / 9.f) : 1.f;
    int total = c.cout * tapsDst * c.cinPad;
    w2bf_kernel<<<(total + 255) / 256, 256, 0, stream>>>(
        (const float*)d_in[c.wI], (ushort_t*)(ws + wbOff[i]),
        c.cin, c.cinPad, tapsSrc, tapsDst, scale, total);
  }

  fc_bank_kernel<<<(256 * 32 * 2048) / 256, 256, 0, stream>>>(
      x2, x3,
      (const float*)d_in[119], (const float*)d_in[114],
      (const float*)d_in[115], (const float*)d_in[116],
      (const float*)d_in[117], (const float*)d_in[118],
      (const float*)d_in[23], (const float*)d_in[18],
      (const float*)d_in[19], (const float*)d_in[20],
      (const float*)d_in[21], (const float*)d_in[22],
      KERN);

  {
    dim3 grid(64, 256);
    x1_transpose_kernel<<<grid, 256, 0, stream>>>(x1, Y1);
  }

  mm_wmma_kernel<<<256, 256, 0, stream>>>(
      KERN, Y1,
      (const float*)d_in[1], (const float*)d_in[0],
      (const float*)d_in[2], (const float*)d_in[3],
      FEAT);

  feat_fill_kernel<<<(16384 * 130 + 255) / 256, 256, 0, stream>>>(x2, x3, FEAT);

  auto conv = [&](int i, const ushort_t* act, ushort_t* outp, int outStride,
                  int coBase, float* pooledp) {
    const CC& c = ccs[i];
    int kchunk = (c.cinPad == 160) ? 160
               : (c.cinPad == 448) ? 224
               : (c.cinPad == 2048) ? 256 : 128;   // 384 -> 128
    dim3 grid(c.cout / 32, 32);
    const float* be = (const float*)d_in[c.bnI];
    const float* g  = (const float*)d_in[c.bnI + 1];
    const float* mu = (const float*)d_in[c.bnI + 2];
    const float* va = (const float*)d_in[c.bnI + 3];
    const ushort_t* wg = (const ushort_t*)(ws + wbOff[i]);
    switch (c.mode) {
      case 0: conv_wmma_kernel<0><<<grid, 256, 0, stream>>>(act, c.cinPad, wg, g, be, mu, va, outp, outStride, coBase, pooledp, kchunk); break;
      case 1: conv_wmma_kernel<1><<<grid, 256, 0, stream>>>(act, c.cinPad, wg, g, be, mu, va, outp, outStride, coBase, pooledp, kchunk); break;
      case 2: conv_wmma_kernel<2><<<grid, 256, 0, stream>>>(act, c.cinPad, wg, g, be, mu, va, outp, outStride, coBase, pooledp, kchunk); break;
      default: conv_wmma_kernel<3><<<grid, 256, 0, stream>>>(act, c.cinPad, wg, g, be, mu, va, outp, outStride, coBase, pooledp, kchunk); break;
    }
  };

  // inception 1: FEAT(160) -> Y1(2048)
  conv(0, FEAT, Y1, 2048, 0,    nullptr);   // b1x1  -> ch 0..319
  conv(1, FEAT, T1, 384,  0,    nullptr);   // b3x3_1
  conv(2, T1,   Y1, 2048, 320,  nullptr);   // b3x3_2a (1x3)
  conv(3, T1,   Y1, 2048, 704,  nullptr);   // b3x3_2b (3x1)
  conv(4, FEAT, T2, 448,  0,    nullptr);   // db1
  conv(5, T2,   T3, 384,  0,    nullptr);   // db2 (3x3)
  conv(6, T3,   Y1, 2048, 1088, nullptr);   // db3a
  conv(7, T3,   Y1, 2048, 1472, nullptr);   // db3b
  conv(8, FEAT, Y1, 2048, 1856, nullptr);   // pool branch (avgpool as 3x3/9)

  // inception 2: Y1(2048) -> pooled [B][2048] (final branches fuse the 8x8 mean)
  conv(9,  Y1, nullptr, 0, 0,    POOL);
  conv(10, Y1, T1, 384, 0,       nullptr);
  conv(11, T1, nullptr, 0, 320,  POOL);
  conv(12, T1, nullptr, 0, 704,  POOL);
  conv(13, Y1, T2, 448, 0,       nullptr);
  conv(14, T2, T3, 384, 0,       nullptr);
  conv(15, T3, nullptr, 0, 1088, POOL);
  conv(16, T3, nullptr, 0, 1472, POOL);
  conv(17, Y1, nullptr, 0, 1856, POOL);

  // FC head
  fc_fused_kernel<<<(256 * 1024 + 255) / 256, 256, 0, stream>>>(
      POOL, (const float*)d_in[9], (const float*)d_in[4],
      (const float*)d_in[6], (const float*)d_in[5],
      (const float*)d_in[7], (const float*)d_in[8],
      H1, 2048, 1024, 256 * 1024, 1);
  fc_fused_kernel<<<(256 * 512 + 255) / 256, 256, 0, stream>>>(
      H1, (const float*)d_in[15], (const float*)d_in[10],
      (const float*)d_in[12], (const float*)d_in[11],
      (const float*)d_in[13], (const float*)d_in[14],
      H2, 1024, 512, 256 * 512, 1);
  fc_fused_kernel<<<(256 * 2 + 255) / 256, 256, 0, stream>>>(
      H2, (const float*)d_in[17], (const float*)d_in[16],
      (const float*)d_in[16], (const float*)d_in[16],
      (const float*)d_in[16], (const float*)d_in[16],
      (float*)d_out, 512, 2, 256 * 2, 0);
}
